// SIGN_22419729285545
// MI455X (gfx1250) — matrix operand order repaired and checked by reading, not compiled, not run on hardware
//
#include <hip/hip_runtime.h>

typedef __attribute__((ext_vector_type(16))) __bf16 v16bf;
typedef __attribute__((ext_vector_type(8)))  __bf16 v8bf;
typedef __attribute__((ext_vector_type(8)))  float  v8f;

#define N_NODES 50000
#define N_EDGES 800000
#define CIN     64
#define HID     256
#define COUT    64
#define WAVES   4

__device__ __forceinline__ unsigned short f2bf(float f) {
    unsigned int u = __float_as_uint(f);
    unsigned int lsb = (u >> 16) & 1u;
    u += 0x7fffu + lsb;                 // round to nearest even
    return (unsigned short)(u >> 16);
}

__device__ __forceinline__ void atomAddF32(float* p, float v) {
    __hip_atomic_fetch_add(p, v, __ATOMIC_RELAXED, __HIP_MEMORY_SCOPE_AGENT);
}

union AFrag { v16bf v; v8bf h[2]; };

// ---------------------------------------------------------------- utilities
__global__ void k_zero(float* __restrict__ p, int n) {
    int i = blockIdx.x * 256 + threadIdx.x;
    if (i < n) p[i] = 0.0f;
}

__global__ void k_degree(const int* __restrict__ col, float* __restrict__ deg, int E) {
    int e = blockIdx.x * 256 + threadIdx.x;
    if (e < E) atomAddF32(&deg[col[e]], 1.0f);
}

__global__ void k_dinv(float* __restrict__ d, int n) {
    int i = blockIdx.x * 256 + threadIdx.x;
    if (i < n) {
        float v = d[i];
        d[i] = (v > 0.0f) ? rsqrtf(v) : 0.0f;
    }
}

// ------------------------------------------------------------------- SpMM
// out[col[e]] += dinv[row[e]]*dinv[col[e]] * h[row[e]]   (16 threads/edge, float4)
__global__ void k_spmm(const float* __restrict__ hin, float* __restrict__ hout,
                       const int* __restrict__ row, const int* __restrict__ col,
                       const float* __restrict__ dinv, int E) {
    int t = blockIdx.x * 256 + threadIdx.x;
    int e = t >> 4;
    if (e >= E) return;
    int c4 = (t & 15) << 2;
    int r  = row[e];
    int d  = col[e];
    float w = dinv[r] * dinv[d];
    if (w == 0.0f) return;
    const float4 hv = *(const float4*)(hin + (size_t)r * CIN + c4);
    float* o = hout + (size_t)d * CIN + c4;
    atomAddF32(o + 0, w * hv.x);
    atomAddF32(o + 1, w * hv.y);
    atomAddF32(o + 2, w * hv.z);
    atomAddF32(o + 3, w * hv.w);
}

// ----------------------------------------------------- pack z = [x|h1|h2|h3] -> bf16
__global__ void k_packz(const float* __restrict__ x,  const float* __restrict__ h1,
                        const float* __restrict__ h2, const float* __restrict__ h3,
                        unsigned short* __restrict__ zb, int n) {
    int t = blockIdx.x * 256 + threadIdx.x;
    if (t >= n * HID) return;
    int i = t >> 8;
    int j = t & 255;
    const float* src = (j < 64) ? x : (j < 128) ? h1 : (j < 192) ? h2 : h3;
    zb[t] = f2bf(src[(size_t)i * CIN + (j & 63)]);
}

// -------------------------------------------- pack W [K][Nout] into B-fragment layout
// per (kTile 32 x nTile 16) block: lane L holds column n = nTile*16 + (L&15),
// K = kTile*32 + (L>=16 ? 8 : 0) + (j&7) + (j>=8 ? 16 : 0), j = 0..15   (ISA B layout for 16-bit WMMA)
__global__ void k_packw(const float* __restrict__ W, unsigned short* __restrict__ out,
                        int K, int Nout) {
    int t = blockIdx.x * 256 + threadIdx.x;
    int nT = Nout >> 4;
    int nTiles = (K >> 5) * nT;
    int lane = t & 31;
    int tile = t >> 5;
    if (tile >= nTiles) return;
    int kTile = tile / nT;
    int nTile = tile % nT;
    int n = nTile * 16 + (lane & 15);
    int kBase = kTile * 32 + ((lane >> 4) << 3);
    unsigned short* dst = out + ((size_t)tile * 32 + lane) * 16;
#pragma unroll
    for (int j = 0; j < 16; ++j)
        dst[j] = f2bf(W[(size_t)(kBase + (j & 7) + ((j >> 3) << 4)) * Nout + n]);
}

// --------------------------------------------------------------- fused MLP (WMMA)
__global__ void __launch_bounds__(32 * WAVES)
k_mlp(const unsigned short* __restrict__ zb,
      const unsigned short* __restrict__ wb0, const float* __restrict__ b0,
      const unsigned short* __restrict__ wb1, const float* __restrict__ b1,
      const unsigned short* __restrict__ wb2, const float* __restrict__ b2,
      float* __restrict__ out, int nNodes) {
    __shared__ __align__(32) unsigned short hbuf[WAVES][2][16 * HID];

    const int wave = threadIdx.x >> 5;
    const int lane = threadIdx.x & 31;
    const int lrow = lane & 15;          // N index inside tile / A row
    const int lhi  = lane >> 4;          // lane half select
    const int colHalf = lhi << 3;        // A fragment K-offset (0 or 8)
    const int m0 = (blockIdx.x * WAVES + wave) * 16;

    AFrag a[8];

    // ---- A fragments of z (global, bf16). ISA 16-bit A layout:
    // lanes 0-15: K = {0..7, 16..23}; lanes 16-31: K = {8..15, 24..31} per 32-K tile.
    {
        int r = m0 + lrow;
        if (r >= nNodes) r = nNodes - 1;     // clamp: keep EXEC all-ones for WMMA
        const unsigned short* zr = zb + (size_t)r * HID;
#pragma unroll
        for (int k = 0; k < 8; ++k) {
            const unsigned short* p = zr + k * 32 + colHalf;
            a[k].h[0] = *(const v8bf*)(p);
            a[k].h[1] = *(const v8bf*)(p + 16);
        }
    }

    // ---- Layer 1: 256 -> 256, relu, stage bf16 in LDS
    unsigned short* h0 = &hbuf[wave][0][0];
#pragma unroll 1
    for (int nt = 0; nt < 16; ++nt) {
        v8f c = {};
#pragma unroll
        for (int k = 0; k < 8; ++k) {
            v16bf b = *(const v16bf*)(wb0 + ((size_t)(k * 16 + nt) * 32 + lane) * 16);
            c = __builtin_amdgcn_wmma_f32_16x16x32_bf16(false, a[k].v, false, b,
                                                        (short)0, c, false, false);
        }
        float bias = b0[nt * 16 + lrow];
#pragma unroll
        for (int r = 0; r < 8; ++r) {
            float v = c[r] + bias;
            v = v > 0.0f ? v : 0.0f;
            h0[(r + (lhi << 3)) * HID + nt * 16 + lrow] = f2bf(v);
        }
    }
    __syncthreads();

    // ---- A fragments from LDS (h0)
#pragma unroll
    for (int k = 0; k < 8; ++k) {
        const unsigned short* p = h0 + lrow * HID + k * 32 + colHalf;
        a[k].h[0] = *(const v8bf*)(p);
        a[k].h[1] = *(const v8bf*)(p + 16);
    }

    // ---- Layer 2: 256 -> 256, relu, stage bf16 in LDS
    unsigned short* h1 = &hbuf[wave][1][0];
#pragma unroll 1
    for (int nt = 0; nt < 16; ++nt) {
        v8f c = {};
#pragma unroll
        for (int k = 0; k < 8; ++k) {
            v16bf b = *(const v16bf*)(wb1 + ((size_t)(k * 16 + nt) * 32 + lane) * 16);
            c = __builtin_amdgcn_wmma_f32_16x16x32_bf16(false, a[k].v, false, b,
                                                        (short)0, c, false, false);
        }
        float bias = b1[nt * 16 + lrow];
#pragma unroll
        for (int r = 0; r < 8; ++r) {
            float v = c[r] + bias;
            v = v > 0.0f ? v : 0.0f;
            h1[(r + (lhi << 3)) * HID + nt * 16 + lrow] = f2bf(v);
        }
    }
    __syncthreads();

#pragma unroll
    for (int k = 0; k < 8; ++k) {
        const unsigned short* p = h1 + lrow * HID + k * 32 + colHalf;
        a[k].h[0] = *(const v8bf*)(p);
        a[k].h[1] = *(const v8bf*)(p + 16);
    }

    // ---- Layer 3: 256 -> 64, bias, store f32 output
#pragma unroll 1
    for (int nt = 0; nt < 4; ++nt) {
        v8f c = {};
#pragma unroll
        for (int k = 0; k < 8; ++k) {
            v16bf b = *(const v16bf*)(wb2 + ((size_t)(k * 4 + nt) * 32 + lane) * 16);
            c = __builtin_amdgcn_wmma_f32_16x16x32_bf16(false, a[k].v, false, b,
                                                        (short)0, c, false, false);
        }
        float bias = b2[nt * 16 + lrow];
#pragma unroll
        for (int r = 0; r < 8; ++r) {
            int node = m0 + r + (lhi << 3);
            if (node < nNodes)
                out[(size_t)node * COUT + nt * 16 + lrow] = c[r] + bias;
        }
    }
}

// ------------------------------------------------------------------ launcher
extern "C" void kernel_launch(void* const* d_in, const int* in_sizes, int n_in,
                              void* d_out, int out_size, void* d_ws, size_t ws_size,
                              hipStream_t stream) {
    const float* x  = (const float*)d_in[0];
    const int*   ei = (const int*)d_in[1];      // edge_index [2][E] (int32 under default JAX)
    const float* W0 = (const float*)d_in[2];
    const float* b0 = (const float*)d_in[3];
    const float* W1 = (const float*)d_in[4];
    const float* b1 = (const float*)d_in[5];
    const float* W2 = (const float*)d_in[6];
    const float* b2 = (const float*)d_in[7];
    float* out = (float*)d_out;

    const int N = N_NODES, E = N_EDGES;
    const int* row = ei;
    const int* col = ei + E;

    // workspace layout (floats first, then bf16/ushort regions)
    const int DINV_PAD = 50176;                          // N rounded up for alignment
    float* ws   = (float*)d_ws;
    float* dinv = ws;                                    // [DINV_PAD]
    float* h1   = ws + DINV_PAD;                         // [N*CIN]
    float* h2   = h1 + (size_t)N * CIN;
    float* h3   = h2 + (size_t)N * CIN;
    unsigned short* zb  = (unsigned short*)(h3 + (size_t)N * CIN);   // [N*HID]
    unsigned short* wb0 = zb + (size_t)N * HID;          // [256*256]
    unsigned short* wb1 = wb0 + 256 * 256;               // [256*256]
    unsigned short* wb2 = wb1 + 256 * 256;               // [256*64]

    size_t needed = (size_t)(DINV_PAD + 3 * N * CIN) * 4 +
                    ((size_t)N * HID + 2 * 256 * 256 + 256 * 64) * 2;
    if (ws_size < needed) return;

    // zero dinv + hop buffers (atomic accumulation targets) every call
    int zeroN = DINV_PAD + 3 * N * CIN;
    k_zero<<<(zeroN + 255) / 256, 256, 0, stream>>>(ws, zeroN);

    k_degree<<<(E + 255) / 256, 256, 0, stream>>>(col, dinv, E);
    k_dinv<<<(N + 255) / 256, 256, 0, stream>>>(dinv, N);

    int spmmT = E * 16;
    k_spmm<<<(spmmT + 255) / 256, 256, 0, stream>>>(x,  h1, row, col, dinv, E);
    k_spmm<<<(spmmT + 255) / 256, 256, 0, stream>>>(h1, h2, row, col, dinv, E);
    k_spmm<<<(spmmT + 255) / 256, 256, 0, stream>>>(h2, h3, row, col, dinv, E);

    int zT = N * HID;
    k_packz<<<(zT + 255) / 256, 256, 0, stream>>>(x, h1, h2, h3, zb, N);

    k_packw<<<((8 * 16 * 32) + 255) / 256, 256, 0, stream>>>(W0, wb0, 256, 256);
    k_packw<<<((8 * 16 * 32) + 255) / 256, 256, 0, stream>>>(W1, wb1, 256, 256);
    k_packw<<<((8 * 4 * 32) + 255) / 256, 256, 0, stream>>>(W2, wb2, 256, 64);

    int tiles  = (N + 15) / 16;                      // 3125 M-tiles
    int blocks = (tiles + WAVES - 1) / WAVES;        // 782
    k_mlp<<<blocks, 32 * WAVES, 0, stream>>>(zb, wb0, b0, wb1, b1, wb2, b2, out, N);
}
